// NemotronRouter_71966472012142
// MI455X (gfx1250) — matrix operand order repaired and checked by reading, not compiled, not run on hardware
//
#include <hip/hip_runtime.h>
#include <hip/hip_bf16.h>

// ---- problem constants (from reference) ----
#define T_DIM    32768
#define D_DIM    2048
#define E_DIM    256
#define NGROUPS  8
#define GSIZE    32      // E_DIM / NGROUPS
#define TOPKG    4
#define TOPK     8
#define SCALE    2.5f

// ---- tiling ----
#define BM        16                 // tokens per workgroup
#define KC        256                // K chunk staged in LDS
#define LDS_HS    (KC + 8)           // bf16 row stride (pad: conflict-free 16B lane reads)
#define SSTRIDE   (E_DIM + 8)        // f32 score row stride

typedef __attribute__((ext_vector_type(8)))  __bf16 v8bf;
typedef __attribute__((ext_vector_type(16))) __bf16 v16bf;
typedef __attribute__((ext_vector_type(8)))  float  v8f;

__device__ __forceinline__ unsigned short f2bf_raw(float f) {
    union { float f; unsigned u; } v; v.f = f;
    unsigned r = v.u + 0x7FFFu + ((v.u >> 16) & 1u);   // round-to-nearest-even
    return (unsigned short)(r >> 16);
}

__device__ __forceinline__ v16bf cat8(v8bf lo, v8bf hi) {
    v16bf r;
#pragma unroll
    for (int i = 0; i < 8; ++i) { r[i] = lo[i]; r[i + 8] = hi[i]; }
    return r;
}

// ---- prologue: W fp32 -> bf16 once (W then lives in L2 as 1 MB) ----
__global__ void wcvt_kernel(const float* __restrict__ W, unsigned short* __restrict__ Wb) {
    const int i = (blockIdx.x * 256 + threadIdx.x) * 8;
    float4 a = ((const float4*)(W + i))[0];
    float4 b = ((const float4*)(W + i))[1];
    union { unsigned short s[8]; uint4 v; } o;
    o.s[0] = f2bf_raw(a.x); o.s[1] = f2bf_raw(a.y);
    o.s[2] = f2bf_raw(a.z); o.s[3] = f2bf_raw(a.w);
    o.s[4] = f2bf_raw(b.x); o.s[5] = f2bf_raw(b.y);
    o.s[6] = f2bf_raw(b.z); o.s[7] = f2bf_raw(b.w);
    *((uint4*)(Wb + i)) = o.v;
}

// ---- fused router: bf16 WMMA GEMM + sigmoid + group top-k ----
__global__ void __launch_bounds__(256) router_kernel(
    const float* __restrict__ H, const unsigned short* __restrict__ Wb,
    const float* __restrict__ bias, int* __restrict__ out_idx,
    float* __restrict__ out_w)
{
    __shared__ unsigned short htile[BM * LDS_HS];   // 16 x (256+8) bf16
    __shared__ float sc[BM * SSTRIDE];              // 16 x 264 f32 scores
    __shared__ float bsh[E_DIM];

    const int tid  = threadIdx.x;
    const int lane = tid & 31;
    const int wave = tid >> 5;
    const int half = (lane >> 4) & 1;
    const int l16  = lane & 15;
    const int tb   = blockIdx.x * BM;

    bsh[tid] = bias[tid];           // 256 threads == E_DIM

    v8f acc0 = {};                  // experts [eb0,    eb0+16)
    v8f acc1 = {};                  // experts [eb0+16, eb0+32)
    const int eb0 = wave * 32;

    const int lrow = tid >> 4;        // 0..15 token row for cooperative load
    const int lcol = (tid & 15) * 16; // 16 fp32 per thread per chunk row

    for (int kc = 0; kc < D_DIM; kc += KC) {
        __syncthreads();
        // stage H chunk into LDS as bf16 (fp32 read once from HBM)
        const float* src = H + (size_t)(tb + lrow) * D_DIM + kc + lcol;
        unsigned short* dst = htile + lrow * LDS_HS + lcol;
#pragma unroll
        for (int i = 0; i < 4; ++i) {
            float4 f = ((const float4*)src)[i];
            union { unsigned short s[4]; uint2 v; } o;
            o.s[0] = f2bf_raw(f.x); o.s[1] = f2bf_raw(f.y);
            o.s[2] = f2bf_raw(f.z); o.s[3] = f2bf_raw(f.w);
            *((uint2*)(dst + i * 4)) = o.v;
        }
        if (kc + KC < D_DIM)
            __builtin_prefetch(src + KC, 0, 1);   // global_prefetch_b8 for next chunk
        __syncthreads();

        const unsigned short* arow = htile + l16 * LDS_HS;
#pragma unroll
        for (int s = 0; s < KC; s += 32) {
            // A frag (16x32 bf16): lane half selects K 0..15 vs 16..31
            v8bf alo = *(const v8bf*)(arow + s + half * 16);
            v8bf ahi = *(const v8bf*)(arow + s + half * 16 + 8);
            v16bf a = cat8(alo, ahi);

            const size_t kg = (size_t)(kc + s) + (size_t)half * 16;
            {   // B frag: lane = column (expert), 16 contiguous K values
                const unsigned short* bp = Wb + (size_t)(eb0 + l16) * D_DIM + kg;
                v16bf b = cat8(*(const v8bf*)bp, *(const v8bf*)(bp + 8));
                acc0 = __builtin_amdgcn_wmma_f32_16x16x32_bf16(
                    false, a, false, b, (short)0, acc0, false, false);
            }
            {
                const unsigned short* bp = Wb + (size_t)(eb0 + 16 + l16) * D_DIM + kg;
                v16bf b = cat8(*(const v8bf*)bp, *(const v8bf*)(bp + 8));
                acc1 = __builtin_amdgcn_wmma_f32_16x16x32_bf16(
                    false, a, false, b, (short)0, acc1, false, false);
            }
        }
    }

    // sigmoid + scatter C tiles into LDS score matrix [token][expert]
#pragma unroll
    for (int v = 0; v < 8; ++v) {
        const int m = v + half * 8;                 // token row in tile
        float s0 = 1.0f / (1.0f + __expf(-acc0[v]));
        float s1 = 1.0f / (1.0f + __expf(-acc1[v]));
        sc[m * SSTRIDE + eb0 + l16]      = s0;
        sc[m * SSTRIDE + eb0 + 16 + l16] = s1;
    }
    __syncthreads();

    // ---- routing: one thread per token (cost is negligible vs GEMM) ----
    if (tid < BM) {
        const int t = tb + tid;
        const float* row = sc + tid * SSTRIDE;

        // group score = sum of top-2 biased scores in each group of 32
        float gsc[NGROUPS];
#pragma unroll
        for (int g = 0; g < NGROUPS; ++g) {
            float m1 = -1e30f, m2 = -1e30f;
            for (int j = 0; j < GSIZE; ++j) {
                float v = row[g * GSIZE + j] + bsh[g * GSIZE + j];
                if (v > m1) { m2 = m1; m1 = v; }
                else if (v > m2) { m2 = v; }
            }
            gsc[g] = m1 + m2;
        }
        // top-4 groups
        unsigned gmask = 0;
#pragma unroll
        for (int r = 0; r < TOPKG; ++r) {
            float best = -1e30f; int bi = 0;
            for (int g = 0; g < NGROUPS; ++g)
                if (!((gmask >> g) & 1u) && gsc[g] > best) { best = gsc[g]; bi = g; }
            gmask |= 1u << bi;
        }
        // top-8 experts within allowed groups (biased scores; all > 0 so
        // masking-to-zero in the reference == restricting the scan here)
        unsigned chosen[E_DIM / 32] = {0, 0, 0, 0, 0, 0, 0, 0};
        int   eidx[TOPK];
        float ew[TOPK];
        float wsum = 0.0f;
        for (int r = 0; r < TOPK; ++r) {
            float best = -1e30f; int bi = 0;
            for (int e = 0; e < E_DIM; ++e) {
                if (!((gmask >> (e >> 5)) & 1u)) continue;
                if ((chosen[e >> 5] >> (e & 31)) & 1u) continue;
                float v = row[e] + bsh[e];
                if (v > best) { best = v; bi = e; }
            }
            chosen[bi >> 5] |= 1u << (bi & 31);
            eidx[r] = bi;
            float s = row[bi];          // un-biased sigmoid score
            ew[r] = s; wsum += s;
        }
        const float f = SCALE / wsum;
#pragma unroll
        for (int r = 0; r < TOPK; ++r) {
            out_idx[(size_t)t * TOPK + r] = eidx[r];
            out_w[(size_t)t * TOPK + r]   = ew[r] * f;
        }
    }
}

extern "C" void kernel_launch(void* const* d_in, const int* in_sizes, int n_in,
                              void* d_out, int out_size, void* d_ws, size_t ws_size,
                              hipStream_t stream) {
    const float* H    = (const float*)d_in[0];   // [T, D] fp32
    const float* W    = (const float*)d_in[1];   // [E, D] fp32
    const float* bias = (const float*)d_in[2];   // [E]    fp32

    unsigned short* Wb = (unsigned short*)d_ws;  // bf16 W, 1 MB of scratch

    int*   out_idx = (int*)d_out;                          // [T, 8] int32
    float* out_w   = ((float*)d_out) + (size_t)T_DIM * TOPK; // [T, 8] f32

    // 1) convert W to bf16 once (2048 elems/block * 256 blocks = 512K elems)
    wcvt_kernel<<<(E_DIM * D_DIM) / (256 * 8), 256, 0, stream>>>(W, Wb);
    // 2) fused GEMM + sigmoid + routing: 16 tokens per block
    router_kernel<<<T_DIM / BM, 256, 0, stream>>>(H, Wb, bias, out_idx, out_w);
}
